// AttentionHead_27238682591463
// MI455X (gfx1250) — matrix operand order repaired and checked by reading, not compiled, not run on hardware
//
#include <hip/hip_runtime.h>
#include <hip/hip_bf16.h>

// Problem constants (match reference)
#define BATCH  8
#define SEQ    2048
#define DMODEL 1024
#define HDIM   64

typedef __attribute__((ext_vector_type(16))) __bf16 v16bf;
typedef __attribute__((ext_vector_type(8)))  __bf16 v8bf;
typedef __attribute__((ext_vector_type(4)))  __bf16 v4bf;
typedef __attribute__((ext_vector_type(8)))  float  v8f;

static __device__ __forceinline__ v8f vzero8() {
  v8f z = {0.f,0.f,0.f,0.f,0.f,0.f,0.f,0.f};
  return z;
}

// D = A(16x32 bf16) x B(32x16 bf16) + C(16x16 f32)
static __device__ __forceinline__ v8f wmma_bf16(v16bf a, v16bf b, v8f c) {
  return __builtin_amdgcn_wmma_f32_16x16x32_bf16(false, a, false, b, (short)0, c,
                                                 false, false);
}

// A-fragment (16xK tile, 32-wide K chunk at kbase), ISA 16-bit A layout:
// lanes 0-15 hold rel-K {0..7,16..23}, lanes 16-31 hold {8..15,24..31}.
static __device__ __forceinline__ v16bf load_a(const __bf16* base, int ld, int kbase) {
  int lane = threadIdx.x & 31;
  int r  = lane & 15;
  int ko = kbase + ((lane >> 4) << 3);
  const __bf16* p = base + (size_t)r * ld + ko;
  v8bf lo = *reinterpret_cast<const v8bf*>(p);
  v8bf hi = *reinterpret_cast<const v8bf*>(p + 16);
  return __builtin_shufflevector(lo, hi, 0,1,2,3,4,5,6,7,8,9,10,11,12,13,14,15);
}

// B-fragment (Kx16 tile, 32-wide K chunk at kbase). base points at the tile's
// first column; column n = lane&15 strides by ld. Per-lane K matches A layout:
// lanes 0-15 rel-K {0..7,16..23}, lanes 16-31 {8..15,24..31}.
static __device__ __forceinline__ v16bf load_b(const __bf16* base, int ld, int kbase) {
  int lane = threadIdx.x & 31;
  int c  = lane & 15;
  int ko = kbase + ((lane >> 4) << 3);
  const __bf16* p = base + (size_t)c * ld + ko;
  v8bf lo = *reinterpret_cast<const v8bf*>(p);
  v8bf hi = *reinterpret_cast<const v8bf*>(p + 16);
  return __builtin_shufflevector(lo, hi, 0,1,2,3,4,5,6,7,8,9,10,11,12,13,14,15);
}

// Reductions across a 16-lane half (C/D-layout rows span lanes 0-15 / 16-31).
static __device__ __forceinline__ float half_max(float v) {
#pragma unroll
  for (int m = 1; m < 16; m <<= 1) v = fmaxf(v, __shfl_xor(v, m, 32));
  return v;
}
static __device__ __forceinline__ float half_sum(float v) {
#pragma unroll
  for (int m = 1; m < 16; m <<= 1) v += __shfl_xor(v, m, 32);
  return v;
}

// ---------------------------------------------------------------------------
// fp32 -> bf16 bulk convert (vectorized; all sizes are multiples of 4)
// ---------------------------------------------------------------------------
__global__ void cvt_f32_to_bf16(const float* __restrict__ in,
                                __bf16* __restrict__ out, int n4) {
  int i = blockIdx.x * blockDim.x + threadIdx.x;
  if (i >= n4) return;
  float4 f = reinterpret_cast<const float4*>(in)[i];
  v4bf r = { (__bf16)f.x, (__bf16)f.y, (__bf16)f.z, (__bf16)f.w };
  reinterpret_cast<v4bf*>(out)[i] = r;
}

// ---------------------------------------------------------------------------
// QKV projection: one wave = 16 rows of [B*S, D] x W^T -> 16x64 (q,k,v).
// q,k stored bf16 [B,S,H]; v stored bf16 transposed [B,H,S].
// ---------------------------------------------------------------------------
__global__ __launch_bounds__(32) void qkv_kernel(
    const __bf16* __restrict__ xb,
    const __bf16* __restrict__ wq, const __bf16* __restrict__ wk,
    const __bf16* __restrict__ wv,
    const float* __restrict__ bq, const float* __restrict__ bk,
    const float* __restrict__ bv,
    __bf16* __restrict__ qb, __bf16* __restrict__ kb, __bf16* __restrict__ vT) {
  const int lane = threadIdx.x & 31;
  const int m0 = blockIdx.x * 16;           // global row tile
  const __bf16* xrow = xb + (size_t)m0 * DMODEL;

  v8f aq[4], ak[4], av[4];
#pragma unroll
  for (int n = 0; n < 4; ++n) { aq[n] = vzero8(); ak[n] = vzero8(); av[n] = vzero8(); }

#pragma unroll 2
  for (int k0 = 0; k0 < DMODEL; k0 += 32) {
    v16bf a = load_a(xrow, DMODEL, k0);
#pragma unroll
    for (int n = 0; n < 4; ++n) {
      v16bf bq_f = load_b(wq + (size_t)(n * 16) * DMODEL, DMODEL, k0);
      aq[n] = wmma_bf16(a, bq_f, aq[n]);
      v16bf bk_f = load_b(wk + (size_t)(n * 16) * DMODEL, DMODEL, k0);
      ak[n] = wmma_bf16(a, bk_f, ak[n]);
      v16bf bv_f = load_b(wv + (size_t)(n * 16) * DMODEL, DMODEL, k0);
      av[n] = wmma_bf16(a, bv_f, av[n]);
    }
  }

  const int hi8 = (lane >> 4) << 3;
  const int cl  = lane & 15;
#pragma unroll
  for (int n = 0; n < 4; ++n) {
    int h = n * 16 + cl;
    float biasq = bq[h], biask = bk[h], biasv = bv[h];
#pragma unroll
    for (int i = 0; i < 8; ++i) {
      int row = m0 + i + hi8;                  // flat b*S+s
      qb[(size_t)row * HDIM + h] = (__bf16)(aq[n][i] + biasq);
      kb[(size_t)row * HDIM + h] = (__bf16)(ak[n][i] + biask);
      int b = row >> 11;                       // /SEQ
      int s = row & (SEQ - 1);
      vT[((size_t)b * HDIM + h) * SEQ + s] = (__bf16)(av[n][i] + biasv);
    }
  }
}

// ---------------------------------------------------------------------------
// Fused causal flash attention + output projection.
// One wave per 16 query rows; streams 32 keys / iteration, online softmax.
// ---------------------------------------------------------------------------
__global__ __launch_bounds__(32) void attn_kernel(
    const __bf16* __restrict__ qb, const __bf16* __restrict__ kb,
    const __bf16* __restrict__ vT, const __bf16* __restrict__ wo,
    const float* __restrict__ bo, float* __restrict__ out) {
  __shared__ __align__(32) __bf16 p_lds[16 * 32];   // P tile (C-layout -> A-layout)
  __shared__ __align__(32) __bf16 o_lds[16 * 64];   // O tile for out-proj

  const int lane  = threadIdx.x & 31;
  const int tiles = SEQ / 16;
  const int b  = blockIdx.x / tiles;
  const int q0 = (blockIdx.x % tiles) * 16;

  const __bf16* qbase = qb + (size_t)b * SEQ * HDIM;
  const __bf16* kbase = kb + (size_t)b * SEQ * HDIM;
  const __bf16* vbase = vT + (size_t)b * HDIM * SEQ;

  // Q fragments (16 rows x 64, two 32-wide K chunks over h)
  v16bf aq0 = load_a(qbase + (size_t)q0 * HDIM, HDIM, 0);
  v16bf aq1 = load_a(qbase + (size_t)q0 * HDIM, HDIM, 32);

  const int hi8 = (lane >> 4) << 3;
  const int cl  = lane & 15;

  float m_[8], l_[8];
  v8f o[4];
#pragma unroll
  for (int i = 0; i < 8; ++i) { m_[i] = -__builtin_inff(); l_[i] = 0.f; }
#pragma unroll
  for (int n = 0; n < 4; ++n) o[n] = vzero8();

  const int kend = q0 + 16;                 // causal: keys <= last row of tile
  const float scale = 0.125f;               // 1/sqrt(64)

  for (int t0 = 0; t0 < kend; t0 += 32) {
    if (t0 + 32 < kend)                     // prefetch next key block (L2)
      __builtin_prefetch(kbase + (size_t)(t0 + 32) * HDIM, 0, 1);

    // scores S = Q K^T  (K-dim = h, two chunks; two 16-key N-tiles)
    v16bf bk00 = load_b(kbase + (size_t)t0 * HDIM, HDIM, 0);
    v16bf bk01 = load_b(kbase + (size_t)t0 * HDIM, HDIM, 32);
    v16bf bk10 = load_b(kbase + (size_t)(t0 + 16) * HDIM, HDIM, 0);
    v16bf bk11 = load_b(kbase + (size_t)(t0 + 16) * HDIM, HDIM, 32);
    v8f s0 = wmma_bf16(aq0, bk00, vzero8());
    s0     = wmma_bf16(aq1, bk01, s0);
    v8f s1 = wmma_bf16(aq0, bk10, vzero8());
    s1     = wmma_bf16(aq1, bk11, s1);

    // mask + online softmax, write P (bf16) to per-wave LDS tile
#pragma unroll
    for (int i = 0; i < 8; ++i) {
      int r = q0 + i + hi8;
      float e0 = (t0 + cl      <= r) ? s0[i] * scale : -__builtin_inff();
      float e1 = (t0 + 16 + cl <= r) ? s1[i] * scale : -__builtin_inff();
      float tm = half_max(fmaxf(e0, e1));
      float mn = fmaxf(m_[i], tm);
      float corr = __expf(m_[i] - mn);
      float p0 = __expf(e0 - mn);
      float p1 = __expf(e1 - mn);
      l_[i] = l_[i] * corr + half_sum(p0 + p1);
      m_[i] = mn;
#pragma unroll
      for (int n = 0; n < 4; ++n) o[n][i] *= corr;
      int rl = i + hi8;
      p_lds[rl * 32 + cl]      = (__bf16)p0;
      p_lds[rl * 32 + 16 + cl] = (__bf16)p1;
    }
    // single wave: DS ops are in-order, compiler preserves the dependency

    // O += P (16x32) x V (32x16 per h-tile); V^T gives contiguous K=t
    v16bf ap = load_a(p_lds, 32, 0);
#pragma unroll
    for (int n = 0; n < 4; ++n) {
      v16bf bvf = load_b(vbase + (size_t)(n * 16) * SEQ + t0, SEQ, 0);
      o[n] = wmma_bf16(ap, bvf, o[n]);
    }
  }

  // normalize rows and stage O to LDS for the fused output projection
#pragma unroll
  for (int i = 0; i < 8; ++i) {
    float inv = 1.f / l_[i];
#pragma unroll
    for (int n = 0; n < 4; ++n) o[n][i] *= inv;
  }
#pragma unroll
  for (int n = 0; n < 4; ++n)
#pragma unroll
    for (int i = 0; i < 8; ++i)
      o_lds[(i + hi8) * 64 + n * 16 + cl] = (__bf16)(o[n][i]);

  // out = O (16x64) x Wo^T (64x64) + bo   (K = h, two chunks; 4 g-tiles)
  v16bf ao0 = load_a(o_lds, 64, 0);
  v16bf ao1 = load_a(o_lds, 64, 32);
#pragma unroll
  for (int g = 0; g < 4; ++g) {
    v16bf bw0 = load_b(wo + (size_t)(g * 16) * HDIM, HDIM, 0);
    v16bf bw1 = load_b(wo + (size_t)(g * 16) * HDIM, HDIM, 32);
    v8f r = wmma_bf16(ao0, bw0, vzero8());
    r     = wmma_bf16(ao1, bw1, r);
    float bias = bo[g * 16 + cl];
#pragma unroll
    for (int i = 0; i < 8; ++i) {
      int row = q0 + i + hi8;
      out[((size_t)b * SEQ + row) * HDIM + g * 16 + cl] = r[i] + bias;
    }
  }
}

// ---------------------------------------------------------------------------
// Host-side launcher
// ---------------------------------------------------------------------------
extern "C" void kernel_launch(void* const* d_in, const int* in_sizes, int n_in,
                              void* d_out, int out_size, void* d_ws, size_t ws_size,
                              hipStream_t stream) {
  (void)in_sizes; (void)n_in; (void)out_size;

  const float* x  = (const float*)d_in[0];
  const float* Wq = (const float*)d_in[1];
  const float* bq = (const float*)d_in[2];
  const float* Wk = (const float*)d_in[3];
  const float* bk = (const float*)d_in[4];
  const float* Wv = (const float*)d_in[5];
  const float* bv = (const float*)d_in[6];
  const float* Wo = (const float*)d_in[7];
  const float* bo = (const float*)d_in[8];
  float* out = (float*)d_out;

  // Workspace layout (bytes); total ~38.4 MB
  char* w = (char*)d_ws;
  const size_t NX  = (size_t)BATCH * SEQ * DMODEL;   // 16,777,216
  const size_t NQ  = (size_t)BATCH * SEQ * HDIM;     //  1,048,576
  const size_t NW  = (size_t)HDIM * DMODEL;          //     65,536
  const size_t NWO = (size_t)HDIM * HDIM;            //      4,096
  __bf16* xb  = (__bf16*)(w);
  __bf16* qb  = (__bf16*)(w + 2 * NX);
  __bf16* kb  = (__bf16*)(w + 2 * (NX + NQ));
  __bf16* vT  = (__bf16*)(w + 2 * (NX + 2 * NQ));
  __bf16* wqb = (__bf16*)(w + 2 * (NX + 3 * NQ));
  __bf16* wkb = (__bf16*)(w + 2 * (NX + 3 * NQ + NW));
  __bf16* wvb = (__bf16*)(w + 2 * (NX + 3 * NQ + 2 * NW));
  __bf16* wob = (__bf16*)(w + 2 * (NX + 3 * NQ + 3 * NW));
  (void)ws_size; // requires >= 2*(NX+3*NQ+3*NW+NWO) ≈ 40.2 MB

  // 1) fp32 -> bf16 conversions
  {
    int n4 = (int)(NX / 4);
    cvt_f32_to_bf16<<<(n4 + 255) / 256, 256, 0, stream>>>(x, xb, n4);
    n4 = (int)(NW / 4);
    cvt_f32_to_bf16<<<(n4 + 255) / 256, 256, 0, stream>>>(Wq, wqb, n4);
    cvt_f32_to_bf16<<<(n4 + 255) / 256, 256, 0, stream>>>(Wk, wkb, n4);
    cvt_f32_to_bf16<<<(n4 + 255) / 256, 256, 0, stream>>>(Wv, wvb, n4);
    n4 = (int)(NWO / 4);
    cvt_f32_to_bf16<<<(n4 + 255) / 256, 256, 0, stream>>>(Wo, wob, n4);
  }

  // 2) QKV projection: 16 rows per wave, 1024 waves
  qkv_kernel<<<(BATCH * SEQ) / 16, 32, 0, stream>>>(xb, wqb, wkb, wvb,
                                                    bq, bk, bv, qb, kb, vT);

  // 3) Fused causal flash attention + output projection
  attn_kernel<<<BATCH * (SEQ / 16), 32, 0, stream>>>(qb, kb, vT, wob, bo, out);
}
